// AutoregressiveRoutingHead_33337536152363
// MI455X (gfx1250) — compile-verified, matrix operand order checked by reading, not run on hardware
//
#include <hip/hip_runtime.h>
#include <hip/hip_bf16.h>

// ---------------------------------------------------------------------------
// GRU decoder head, MI455X / gfx1250, bf16 WMMA (16x16x32) with f32 accum.
//
//  B=65536, L=8, LATENT=256, HID=128, NTOK=5
//  - gi = emb[token]@W_ih^T collapses to a 6x768 table G (teacher forcing).
//  - r/z gates: G fused into the recurrent GEMM via one-hot token columns
//    appended to h (K = 256 -> 288).
//  - n gate: gh_n from GEMM (K=256 only), i_n from LDS table, b_hh_n inside
//    the r-product (PyTorch GRU semantics).
//  - h double-buffered in LDS: step t reads buf[t&1], writes buf[t&1^1].
//  - Gate j-dim in two 128-col halves; W staged 256 rows/stage, 3 stages/step.
//  - Rounds 3-6: compiler kept B-tile VALUES live across the gate and spilled
//    them to scratch regardless of fences/caps. Fix: operand tiles are loaded
//    by asm-volatile ds_load_b128 with an explicit 2-tile software pipeline
//    and exact s_wait_dscnt immediates (DS retires in FIFO order). The wait
//    asm passes the consumed tiles through "+v" so copies land after the wait.
//  - logits: one padded 16x16 WMMA N-tile per wave per step.
// ---------------------------------------------------------------------------

typedef __attribute__((ext_vector_type(16))) __bf16    v16bf;
typedef __attribute__((ext_vector_type(8)))  float     v8f;
typedef __attribute__((ext_vector_type(4)))  unsigned  u32x4;

#if __has_builtin(__builtin_amdgcn_sched_barrier)
#define SCHED_FENCE() __builtin_amdgcn_sched_barrier(0)
#else
#define SCHED_FENCE()
#endif

#if __has_attribute(amdgpu_num_vgpr)
#define VGPR_CAP __attribute__((amdgpu_num_vgpr(256)))
#else
#define VGPR_CAP
#endif

#define SEQ_L   8
#define LATENT  256
#define HID     128
#define NTOK    5
#define WX_ROW  288          // packed global row (bf16): 256 + 32 one-hot/pad
#define ROWS_WG 128
#define THREADS 256

// LDS layout (bytes). Row stride 592 B => conflict-free b128 tiles.
#define H_STRIDE 592
#define H0_OFF   0
#define H1_OFF   (128 * 592)                 // 75776
#define W_OFF    (2 * 128 * 592)             // 151552; staged W: 256 rows
#define W_SIZE   (256 * 592)                 // 151552
#define GN_OFF   (W_OFF + W_SIZE)            // 303104; G_n f32 [6][256]
#define GN_SIZE  (6 * 256 * 4)
#define WOB_OFF  (GN_OFF + GN_SIZE)          // 309248; W_out bf16 16 rows
#define WOB_SIZE (16 * 592)
#define BOUT_OFF (WOB_OFF + WOB_SIZE)        // 318720
#define BOUT_SIZE 32
#define BHH_OFF  (BOUT_OFF + BOUT_SIZE)      // 318752; b_hh n-slice f32[256]
#define BHH_SIZE (256 * 4)
#define TOK_OFF  (BHH_OFF + BHH_SIZE)        // 319776; tokens int [128][8]
#define TOK_SIZE (128 * 8 * 4)
#define SMEM_BYTES (TOK_OFF + TOK_SIZE)      // 323872 <= 327680 (320KB)

// workspace layout (bytes)
#define WS_WX_OFF   0                        // bf16 [768][288] = 442368
#define WS_GN_OFF   442368                   // f32  [6][256]   = 6144
#define WS_WOUT_OFF 448512                   // bf16 [16][256]  = 8192

union TileU { u32x4 q[2]; v16bf v; };

// ---- asm tile pipeline ------------------------------------------------------
// Issue one 16x32 bf16 operand tile (two ds_load_b128). Volatile: cannot be
// CSE'd / hoisted / clause-merged; issue order is program order.
__device__ __forceinline__ void ds_issue(TileU& d, const char* laneBase, int imm) {
    unsigned addr = (unsigned)(size_t)laneBase;   // low 32 bits = LDS offset
    asm volatile("ds_load_b128 %0, %2 offset:%3\n"
                 "ds_load_b128 %1, %2 offset:%4"
                 : "=v"(d.q[0]), "=v"(d.q[1])
                 : "v"(addr), "n"(imm), "n"(imm + 32));
}

// Wait until at most `imm` DS ops outstanding; ties the consumed tiles through
// the asm so any compiler copies are placed after the wait.
__device__ __forceinline__ void ds_wait_tiles(int imm, TileU& b, TileU& a) {
    asm volatile("s_wait_dscnt %4"
                 : "+v"(b.q[0]), "+v"(b.q[1]), "+v"(a.q[0]), "+v"(a.q[1])
                 : "n"(imm));
}

// Plain tile load (used only for the tiny logits GEMM).
__device__ __forceinline__ v16bf lds_tile(const char* laneBase, int imm) {
    TileU u;
    u.q[0] = *(const u32x4*)(laneBase + imm);
    u.q[1] = *(const u32x4*)(laneBase + imm + 32);
    return u.v;
}

__device__ __forceinline__ void write_tail(char* hbuf, int row, int tk) {
    // one-hot token block (K 256..287) of an h row, bf16(1.0) = 0x3F80
    u32x4 q0 = {0u, 0u, 0u, 0u};
    u32x4 q1 = {0u, 0u, 0u, 0u};
    unsigned val = (tk & 1) ? 0x3F800000u : 0x00003F80u;
    q0[(unsigned)tk >> 1] = val;             // tk in 0..5 -> slot 0..2
    u32x4* dst = (u32x4*)(hbuf + row * H_STRIDE + 512);
    dst[0] = q0;
    dst[1] = q1;
}

// Stage two 128-row gate blocks into W slots 0/1.
__device__ __forceinline__ void stage2(char* smem, const __bf16* __restrict__ Wx,
                                       int jgTop, int jgBot, int tid) {
    for (int i = tid; i < 256 * 36; i += THREADS) {
        int r = i / 36, cb = i - r * 36;
        int jg = (r < 128) ? (jgTop + r) : (jgBot + (r - 128));
        *(u32x4*)(smem + W_OFF + r * H_STRIDE + cb * 16) =
            ((const u32x4*)(Wx + (size_t)jg * WX_ROW))[cb];
    }
}

__device__ __forceinline__ void prefetch2(const __bf16* __restrict__ Wx,
                                          int jgTop, int jgBot, int tid) {
    for (int i = tid; i < 1152; i += THREADS) {
        int jg = (i < 576) ? jgTop : jgBot;
        int l  = (i < 576) ? i : i - 576;
        __builtin_prefetch((const char*)(Wx + (size_t)jg * WX_ROW) + l * 128);
    }
}

// One gate GEMM over 8 N-tiles (kt-outer, acc index = g&7 so consecutive
// WMMAs hit different accumulators). 2-tile lookahead; exact FIFO dscnt waits.
template <int KT>
__device__ __forceinline__ void sweep(const char* aBase, const char* wSlot, v8f acc[8]) {
    const char* wHi = wSlot + 4 * 16 * H_STRIDE;   // tiles 4..7 (imm < 64KB)
    TileU A[2], B[3];
    const int TOT = KT * 8;

    auto issue = [&](int g) {
        int kt = g >> 3, tl = g & 7;
        if (tl == 0) ds_issue(A[kt & 1], aBase, kt * 64);
        ds_issue(B[g % 3], (tl < 4) ? wSlot : wHi, (tl & 3) * 16 * H_STRIDE + kt * 64);
    };

    issue(0);
    issue(1);
#pragma unroll
    for (int g = 0; g < TOT; ++g) {
        if (g + 2 < TOT) issue(g + 2);
        // DS ops issued after tile g's pair: tiles g+1..last (2 each) plus the
        // A-load accompanying any slice start among them (2 each).
        int last = (g + 2 < TOT) ? (g + 2) : (TOT - 1);
        int imm = 2 * (last - g);
#pragma unroll
        for (int x = g + 1; x <= g + 2; ++x)
            if (x <= last && (x & 7) == 0) imm += 2;
        int kt = g >> 3;
        ds_wait_tiles(imm, B[g % 3], A[kt & 1]);
        acc[g & 7] = __builtin_amdgcn_wmma_f32_16x16x32_bf16(
            false, A[kt & 1].v, false, B[g % 3].v, (short)0, acc[g & 7], false, false);
    }
}

// Fast activations on the transcendental pipe (co-executes with XDL WMMA).
__device__ __forceinline__ float sigmoid_fast(float x) {
    return __builtin_amdgcn_rcpf(1.0f + __expf(-x));
}
__device__ __forceinline__ float tanh_fast(float x) {
    return 1.0f - 2.0f * __builtin_amdgcn_rcpf(1.0f + __expf(2.0f * x));
}

// ---------------------------------------------------------------------------
// Prep kernel: Wx (bf16 [768][288] = [W_hh | fused G one-hot cols]),
// G_n table (f32 [6][256], x-part + b_ih only), padded W_out (bf16 [16][256]).
// ---------------------------------------------------------------------------
extern "C" __global__ void gru_prep(const float* __restrict__ emb,
                                    const float* __restrict__ W_ih,
                                    const float* __restrict__ W_hh,
                                    const float* __restrict__ b_ih,
                                    const float* __restrict__ b_hh,
                                    const float* __restrict__ W_out,
                                    __bf16* __restrict__ Wx,
                                    float* __restrict__ Gn,
                                    __bf16* __restrict__ WoutX) {
    const int j = blockIdx.x;       // 0..767
    const int tid = threadIdx.x;    // 0..255

    Wx[j * WX_ROW + tid] = (__bf16)W_hh[j * LATENT + tid];

    if (tid < 32) {
        float v = 0.0f;
        if (tid < 6) {
            float s = b_ih[j] + ((j < 512) ? b_hh[j] : 0.0f);
            for (int k = 0; k < HID; ++k)
                s += emb[tid * HID + k] * W_ih[j * HID + k];
            if (j >= 512) Gn[tid * 256 + (j - 512)] = s;   // n-gate: kept separate
            v = (j < 512) ? s : 0.0f;                      // r/z: fused into GEMM
        }
        Wx[j * WX_ROW + 256 + tid] = (__bf16)v;
    }
    if (j < 16) {
        WoutX[j * 256 + tid] = (__bf16)((j < NTOK) ? W_out[j * 256 + tid] : 0.0f);
    }
}

// ---------------------------------------------------------------------------
// Main kernel: 256 threads = 8 waves, 128 batch rows per WG (16 per wave).
// ---------------------------------------------------------------------------
extern "C" __global__ void __launch_bounds__(THREADS) VGPR_CAP
gru_main(const float* __restrict__ latent, const int* __restrict__ target,
         const __bf16* __restrict__ Wx, const float* __restrict__ GnG,
         const __bf16* __restrict__ WoutX, const float* __restrict__ b_out,
         const float* __restrict__ b_hh, float* __restrict__ out) {
    extern __shared__ char smem[];
    const int tid  = threadIdx.x;
    const int lane = tid & 31, wave = tid >> 5;
    const int ln = lane & 15, hi = lane >> 4;
    const int wr0 = wave * 16;
    const long grow0 = (long)blockIdx.x * ROWS_WG;

    float* gn   = (float*)(smem + GN_OFF);
    float* bout = (float*)(smem + BOUT_OFF);
    int*   tok  = (int*)(smem + TOK_OFF);

    // ---- one-time staging ----
    for (int i = tid; i < 6 * 256; i += THREADS) gn[i] = GnG[i];
    for (int i = tid; i < 256; i += THREADS) ((float*)(smem + BHH_OFF))[i] = b_hh[512 + i];
    if (tid < 8) bout[tid] = (tid < NTOK) ? b_out[tid] : 0.0f;
    for (int i = tid; i < 16 * 32; i += THREADS) {               // W_out 16B chunks
        int r = i >> 5, cb = i & 31;
        *(u32x4*)(smem + WOB_OFF + r * H_STRIDE + cb * 16) = ((const u32x4*)WoutX)[r * 32 + cb];
    }
    for (int i = tid; i < 128 * SEQ_L; i += THREADS) {           // teacher-forced tokens
        int row = i >> 3, t = i & 7;
        tok[i] = (t == 0) ? NTOK : target[(grow0 + row) * SEQ_L + (t - 1)];
    }
    for (int i = tid; i < 128 * LATENT; i += THREADS) {          // h0 -> buffer 0
        int row = i >> 8, c = i & 255;
        ((__bf16*)(smem + H0_OFF))[row * 296 + c] = (__bf16)latent[(grow0 + row) * LATENT + c];
    }
    for (int i = tid; i < 128; i += THREADS) write_tail(smem + H0_OFF, i, NTOK);
    __syncthreads();

    // ---- hoisted per-lane LDS bases (all later offsets are immediates) ----
    const char* aB[2]   = { smem + H0_OFF + (wr0 + ln) * H_STRIDE + hi * 16,
                            smem + H1_OFF + (wr0 + ln) * H_STRIDE + hi * 16 };
    const char* rowB[2] = { smem + H0_OFF + (wr0 + 8 * hi) * H_STRIDE + ln * 2,
                            smem + H1_OFF + (wr0 + 8 * hi) * H_STRIDE + ln * 2 };
    const char* wS0  = smem + W_OFF + ln * H_STRIDE + hi * 16;
    const char* wS1  = wS0 + 128 * H_STRIDE;
    const char* wobB = smem + WOB_OFF + ln * H_STRIDE + hi * 16;
    const float* bhhL = (const float*)(smem + BHH_OFF) + ln;
    const int*   tokL = (const int*)(smem + TOK_OFF) + (wr0 + 8 * hi) * SEQ_L;

    const v8f z8 = {0.f, 0.f, 0.f, 0.f, 0.f, 0.f, 0.f, 0.f};

#pragma unroll 1
    for (int t = 0; t < SEQ_L; ++t) {
        const int cur = t & 1, nx = cur ^ 1;
        const char* aC = aB[cur];
        const char* hOldRow = rowB[cur];
        char*       hNewRow = (char*)rowB[nx];
        char*       hNewBuf = (char*)smem + (nx ? H1_OFF : H0_OFF);

        int tkv[8];
#pragma unroll
        for (int e = 0; e < 8; ++e) tkv[e] = tokL[e * SEQ_L + t];

        v8f r8[8], n8[8];

        auto gate_r = [&](const char* wSlot) {
            v8f acc[8];
#pragma unroll
            for (int tl = 0; tl < 8; ++tl) acc[tl] = z8;
            sweep<9>(aC, wSlot, acc);
#pragma unroll
            for (int tl = 0; tl < 8; ++tl)
#pragma unroll
                for (int e = 0; e < 8; ++e)
                    r8[tl][e] = sigmoid_fast(acc[tl][e]);
        };
        auto gate_n = [&](const char* wSlot, int jb) {
            v8f acc[8];
#pragma unroll
            for (int tl = 0; tl < 8; ++tl) acc[tl] = z8;
            sweep<8>(aC, wSlot, acc);                       // pure-h K range
#pragma unroll
            for (int tl = 0; tl < 8; ++tl)
#pragma unroll
                for (int e = 0; e < 8; ++e) {
                    float gi_n = *((const float*)(smem + GN_OFF) + tkv[e] * 256 + jb + tl * 16 + ln);
                    float gh_n = acc[tl][e] + bhhL[jb + tl * 16];
                    n8[tl][e] = tanh_fast(gi_n + r8[tl][e] * gh_n);
                }
        };
        auto gate_z_commit = [&](const char* wSlot, int jb) {
            v8f acc[8];
#pragma unroll
            for (int tl = 0; tl < 8; ++tl) acc[tl] = z8;
            sweep<9>(aC, wSlot, acc);
#pragma unroll
            for (int tl = 0; tl < 8; ++tl)
#pragma unroll
                for (int e = 0; e < 8; ++e) {
                    float z = sigmoid_fast(acc[tl][e]);
                    float hold = (float)*(const __bf16*)(hOldRow + e * H_STRIDE + jb * 2 + tl * 32);
                    float hnew = z * hold + (1.0f - z) * n8[tl][e];
                    *(__bf16*)(hNewRow + e * H_STRIDE + jb * 2 + tl * 32) = (__bf16)hnew;
                }
        };

        // ---- stage 0: r.h0 | n.h0 ----
        stage2(smem, Wx, 0, 512, tid);
        __syncthreads();
        prefetch2(Wx, 256, 128, tid);
        SCHED_FENCE();
        gate_r(wS0);
        SCHED_FENCE();
        gate_n(wS1, 0);
        SCHED_FENCE();
        __syncthreads();

        // ---- stage 1: z.h0 | r.h1 ----
        stage2(smem, Wx, 256, 128, tid);
        __syncthreads();
        prefetch2(Wx, 640, 384, tid);
        SCHED_FENCE();
        gate_z_commit(wS0, 0);
        SCHED_FENCE();
        gate_r(wS1);
        SCHED_FENCE();
        __syncthreads();

        // ---- stage 2: n.h1 | z.h1 ----
        stage2(smem, Wx, 640, 384, tid);
        __syncthreads();
        if (t + 1 < SEQ_L) prefetch2(Wx, 0, 512, tid);
        SCHED_FENCE();
        gate_n(wS0, 128);
        SCHED_FENCE();
        gate_z_commit(wS1, 128);
        SCHED_FENCE();
        __syncthreads();

        // ---- one-hot tail for next step's token (into the new h buffer) ----
        if (lane < 16 && (t + 1) < SEQ_L)
            write_tail(hNewBuf, wr0 + lane, tokL[(lane - 8 * hi) * SEQ_L + (t + 1)]);
        SCHED_FENCE();

        // ---- logits: h_new (16x256) @ W_out^T (N padded to 16), 8 WMMA ----
        {
            const char* aN = aB[nx];
            v8f acc = z8;
#pragma unroll
            for (int kt = 0; kt < 8; ++kt) {
                v16bf a = lds_tile(aN, kt * 64);
                v16bf b = lds_tile(wobB, kt * 64);
                acc = __builtin_amdgcn_wmma_f32_16x16x32_bf16(
                    false, a, false, b, (short)0, acc, false, false);
                SCHED_FENCE();
            }
            if (ln < NTOK) {
#pragma unroll
                for (int e = 0; e < 8; ++e) {
                    long row = grow0 + wr0 + e + 8 * hi;
                    out[(row * SEQ_L + t) * NTOK + ln] = acc[e] + bout[ln];
                }
            }
        }
        SCHED_FENCE();
        // no extra barrier: W readers synced above; h buffers are wave-private
    }
}

// ---------------------------------------------------------------------------
extern "C" void kernel_launch(void* const* d_in, const int* in_sizes, int n_in,
                              void* d_out, int out_size, void* d_ws, size_t ws_size,
                              hipStream_t stream) {
    const float* latent = (const float*)d_in[0];
    const int*   target = (const int*)d_in[1];
    const float* emb    = (const float*)d_in[2];
    const float* W_ih   = (const float*)d_in[3];
    const float* W_hh   = (const float*)d_in[4];
    const float* b_ih   = (const float*)d_in[5];
    const float* b_hh   = (const float*)d_in[6];
    const float* W_out  = (const float*)d_in[7];
    const float* b_out  = (const float*)d_in[8];
    float* out = (float*)d_out;

    char* ws = (char*)d_ws;
    __bf16* Wx    = (__bf16*)(ws + WS_WX_OFF);
    float*  Gn    = (float*)(ws + WS_GN_OFF);
    __bf16* WoutX = (__bf16*)(ws + WS_WOUT_OFF);

    hipFuncSetAttribute((const void*)gru_main,
                        hipFuncAttributeMaxDynamicSharedMemorySize, SMEM_BYTES);

    gru_prep<<<768, 256, 0, stream>>>(emb, W_ih, W_hh, b_ih, b_hh, W_out, Wx, Gn, WoutX);
    gru_main<<<65536 / ROWS_WG, THREADS, SMEM_BYTES, stream>>>(
        latent, target, Wx, Gn, WoutX, b_out, b_hh, out);
}
